// MultiHeadAttention_49409303773925
// MI455X (gfx1250) — matrix operand order repaired and checked by reading, not compiled, not run on hardware
//
#include <hip/hip_runtime.h>
#include <hip/hip_bf16.h>

// ---------------------------------------------------------------------------
// MHA with relative terms, for MI455X (gfx1250, wave32, WMMA).
// Dims: B=4, L=128, H=8, d=128, model=1024. All reshapes are flat reinterprets.
// GEMMs run on v_wmma_f32_16x16x32_bf16 (f32 accumulate); the 536MB relative-
// tensor streams use nontemporal vector loads (memory-bound part).
// ---------------------------------------------------------------------------

typedef __attribute__((ext_vector_type(16))) __bf16 v16bf;
typedef __attribute__((ext_vector_type(8)))  float  v8f;
typedef __attribute__((ext_vector_type(4)))  float  f32x4;

__device__ __forceinline__ unsigned short f32_to_bf16(float f) {
  unsigned u = __float_as_uint(f);
  u += 0x7FFFu + ((u >> 16) & 1u);      // round-to-nearest-even
  return (unsigned short)(u >> 16);
}
__device__ __forceinline__ unsigned pack_bf16x2(float a, float b) {
  return (unsigned)f32_to_bf16(a) | ((unsigned)f32_to_bf16(b) << 16);
}
// A-fragment K interleave for 16-bit WMMA (ISA 7.12.2): per half-lane the 16
// stored halves correspond to K = {0..7,16..23} / {8..15,24..31}.
__device__ __forceinline__ int kperm(int k) {
  int g = k >> 3;
  return (g == 1) ? (k + 8) : ((g == 2) ? (k - 8) : k);
}

// ---------------------------------------------------------------------------
// C[M,N] = A[M,K] * Bm[N,K]^T (+bias[N]) (+res[M,N]); batched via blockIdx.z.
// 128 threads = 4 waves; 64x64 tile; K-step 32.
// ---------------------------------------------------------------------------
__global__ __launch_bounds__(128)
void gemm_xwT_bf16(const float* __restrict__ A, const float* __restrict__ Bm,
                   const float* __restrict__ bias, const float* __restrict__ res,
                   float* __restrict__ C, int M, int N, int K,
                   long long sA, long long sB, long long sC) {
  __shared__ __align__(32) unsigned short As[64 * 32];
  __shared__ __align__(32) unsigned short Bs[64 * 32];
  const int tid = threadIdx.x, lane = tid & 31, wv = tid >> 5;
  const int n0 = blockIdx.x * 64, m0 = blockIdx.y * 64;
  A += blockIdx.z * sA; Bm += blockIdx.z * sB; C += blockIdx.z * sC;

  v8f acc[4] = {};
  const int r = tid >> 1, cb = (tid & 1) << 4;

  for (int kt = 0; kt < K; kt += 32) {
    __syncthreads();
    // Stage A (rows m0.., linear K) and B (W rows n0.., linear K) as bf16.
    const float* srcA = A + (long long)(m0 + r) * K + kt + cb;
    const float* srcB = Bm + (long long)(n0 + r) * K + kt + cb;
#pragma unroll
    for (int c4 = 0; c4 < 4; ++c4) {
      f32x4 va = *(const f32x4*)(srcA + c4 * 4);
      f32x4 vb = *(const f32x4*)(srcB + c4 * 4);
      int c = cb + c4 * 4;
      *(unsigned*)&As[r * 32 + c]      = pack_bf16x2(va.x, va.y);
      *(unsigned*)&As[r * 32 + c + 2]  = pack_bf16x2(va.z, va.w);
      *(unsigned*)&Bs[r * 32 + c]      = pack_bf16x2(vb.x, vb.y);
      *(unsigned*)&Bs[r * 32 + c + 2]  = pack_bf16x2(vb.z, vb.w);
    }
    __syncthreads();

    const int row = lane & 15, half = lane >> 4;
    v16bf a = *(const v16bf*)&As[(wv * 16 + row) * 32 + half * 16];
#pragma unroll
    for (int t = 0; t < 4; ++t) {
      v16bf b = *(const v16bf*)&Bs[(t * 16 + row) * 32 + half * 16];
      acc[t] = __builtin_amdgcn_wmma_f32_16x16x32_bf16(
          false, a, false, b, (short)0, acc[t], false, false);
    }
  }

  const int half = lane >> 4, nn = lane & 15;
#pragma unroll
  for (int t = 0; t < 4; ++t) {
#pragma unroll
    for (int rr = 0; rr < 8; ++rr) {
      int gm = m0 + wv * 16 + half * 8 + rr;
      int gn = n0 + t * 16 + nn;
      float v = acc[t][rr];
      if (bias) v += bias[gn];
      if (res)  v += res[(long long)gm * N + gn];
      C[(long long)gm * N + gn] = v;
    }
  }
}

// ---------------------------------------------------------------------------
// C[M,N] = A[M,K] * Bm[K,N]  (B transposed through LDS); batched via z.
// ---------------------------------------------------------------------------
__global__ __launch_bounds__(128)
void gemm_ab_bf16(const float* __restrict__ A, const float* __restrict__ Bm,
                  float* __restrict__ C, int M, int N, int K,
                  long long sA, long long sB, long long sC) {
  __shared__ __align__(32) unsigned short As[64 * 32];
  __shared__ __align__(32) unsigned short Bs[64 * 32];
  const int tid = threadIdx.x, lane = tid & 31, wv = tid >> 5;
  const int n0 = blockIdx.x * 64, m0 = blockIdx.y * 64;
  A += blockIdx.z * sA; Bm += blockIdx.z * sB; C += blockIdx.z * sC;

  v8f acc[4] = {};
  const int ra = tid >> 1, cba = (tid & 1) << 4;  // A staging
  const int kr = tid >> 2, nb = (tid & 3) << 4;   // B staging (transpose)

  for (int kt = 0; kt < K; kt += 32) {
    __syncthreads();
    const float* srcA = A + (long long)(m0 + ra) * K + kt + cba;
    const float* srcB = Bm + (long long)(kt + kr) * N + n0 + nb;
#pragma unroll
    for (int c4 = 0; c4 < 4; ++c4) {
      f32x4 va = *(const f32x4*)(srcA + c4 * 4);
      int c = cba + c4 * 4;
      *(unsigned*)&As[ra * 32 + c]     = pack_bf16x2(va.x, va.y);
      *(unsigned*)&As[ra * 32 + c + 2] = pack_bf16x2(va.z, va.w);
      f32x4 vb = *(const f32x4*)(srcB + c4 * 4);
      int n = nb + c4 * 4;
      Bs[(n + 0) * 32 + kr] = f32_to_bf16(vb.x);  // B linear in K per column
      Bs[(n + 1) * 32 + kr] = f32_to_bf16(vb.y);
      Bs[(n + 2) * 32 + kr] = f32_to_bf16(vb.z);
      Bs[(n + 3) * 32 + kr] = f32_to_bf16(vb.w);
    }
    __syncthreads();

    const int row = lane & 15, half = lane >> 4;
    v16bf a = *(const v16bf*)&As[(wv * 16 + row) * 32 + half * 16];
#pragma unroll
    for (int t = 0; t < 4; ++t) {
      v16bf b = *(const v16bf*)&Bs[(t * 16 + row) * 32 + half * 16];
      acc[t] = __builtin_amdgcn_wmma_f32_16x16x32_bf16(
          false, a, false, b, (short)0, acc[t], false, false);
    }
  }

  const int half = lane >> 4, nn = lane & 15;
#pragma unroll
  for (int t = 0; t < 4; ++t)
#pragma unroll
    for (int rr = 0; rr < 8; ++rr) {
      int gm = m0 + wv * 16 + half * 8 + rr;
      int gn = n0 + t * 16 + nn;
      C[(long long)gm * N + gn] = acc[t][rr];
    }
}

// ---------------------------------------------------------------------------
// attn[bh,i,j] = softmax_j( (S[bh,i,j] + dot(q[bh,i,:], relq[bh,i,j,:])) * 0.25 )
// One block per (bh,i); 128 threads, thread j streams its 512B rel chunk NT.
// ---------------------------------------------------------------------------
__global__ __launch_bounds__(128)
void rel_attn_softmax(const float* __restrict__ q, const float* __restrict__ relq,
                      float* __restrict__ attn) {
  __shared__ float qrow[128];
  __shared__ float red[128];
  const int i = blockIdx.x, bh = blockIdx.y, j = threadIdx.x;
  const long long rowbase = ((long long)bh * 128 + i) * 128;
  qrow[j] = q[rowbase + j];
  __syncthreads();

  const float* rp = relq + (rowbase + j) * 128;
  float s = 0.f;
#pragma unroll 8
  for (int d = 0; d < 128; d += 4) {
    f32x4 r4 = __builtin_nontemporal_load((const f32x4*)(rp + d));
    s += qrow[d] * r4.x + qrow[d + 1] * r4.y + qrow[d + 2] * r4.z + qrow[d + 3] * r4.w;
  }
  float logit = (attn[rowbase + j] + s) * 0.25f;  // scale = (128//8)^-0.5

  red[j] = logit; __syncthreads();
  for (int st = 64; st > 0; st >>= 1) {
    if (j < st) red[j] = fmaxf(red[j], red[j + st]);
    __syncthreads();
  }
  float m = red[0]; __syncthreads();
  float e = __expf(logit - m);
  red[j] = e; __syncthreads();
  for (int st = 64; st > 0; st >>= 1) {
    if (j < st) red[j] += red[j + st];
    __syncthreads();
  }
  attn[rowbase + j] = e * (1.0f / red[0]);
}

// ---------------------------------------------------------------------------
// vnew[bh,i,j] = v[bh,i,j] + dot(v[bh,i,:], relv[bh,i,j,:])   (L == d)
// ---------------------------------------------------------------------------
__global__ __launch_bounds__(128)
void rel_v_update(const float* __restrict__ v, const float* __restrict__ relv,
                  float* __restrict__ vnew) {
  __shared__ float vrow[128];
  const int i = blockIdx.x, bh = blockIdx.y, j = threadIdx.x;
  const long long rowbase = ((long long)bh * 128 + i) * 128;
  vrow[j] = v[rowbase + j];
  __syncthreads();
  const float* rp = relv + (rowbase + j) * 128;
  float s = 0.f;
#pragma unroll 8
  for (int d = 0; d < 128; d += 4) {
    f32x4 r4 = __builtin_nontemporal_load((const f32x4*)(rp + d));
    s += vrow[d] * r4.x + vrow[d + 1] * r4.y + vrow[d + 2] * r4.z + vrow[d + 3] * r4.w;
  }
  vnew[rowbase + j] = vrow[j] + s;
}

// ---------------------------------------------------------------------------
// LayerNorm over last dim (1024) per row; 256 threads per row.
// ---------------------------------------------------------------------------
__global__ __launch_bounds__(256)
void layernorm_1024(const float* __restrict__ x, const float* __restrict__ gamma,
                    const float* __restrict__ beta, float* __restrict__ out) {
  __shared__ float rs[256], rs2[256];
  const int row = blockIdx.x, tid = threadIdx.x;
  const float* xr = x + (long long)row * 1024;
  float vals[4], s = 0.f, s2 = 0.f;
#pragma unroll
  for (int t = 0; t < 4; ++t) {
    float xv = xr[tid + t * 256];
    vals[t] = xv; s += xv; s2 += xv * xv;
  }
  rs[tid] = s; rs2[tid] = s2; __syncthreads();
  for (int st = 128; st > 0; st >>= 1) {
    if (tid < st) { rs[tid] += rs[tid + st]; rs2[tid] += rs2[tid + st]; }
    __syncthreads();
  }
  float mu = rs[0] * (1.0f / 1024.0f);
  float var = rs2[0] * (1.0f / 1024.0f) - mu * mu;
  float rstd = rsqrtf(var + 1e-5f);
#pragma unroll
  for (int t = 0; t < 4; ++t) {
    int c = tid + t * 256;
    out[(long long)row * 1024 + c] = (vals[t] - mu) * rstd * gamma[c] + beta[c];
  }
}

// ---------------------------------------------------------------------------
extern "C" void kernel_launch(void* const* d_in, const int* in_sizes, int n_in,
                              void* d_out, int out_size, void* d_ws, size_t ws_size,
                              hipStream_t stream) {
  const float* key   = (const float*)d_in[0];
  const float* value = (const float*)d_in[1];
  const float* query = (const float*)d_in[2];
  const float* relq  = (const float*)d_in[3];
  const float* relv  = (const float*)d_in[4];
  const float* Wq = (const float*)d_in[5];  const float* bq = (const float*)d_in[6];
  const float* Wk = (const float*)d_in[7];  const float* bk = (const float*)d_in[8];
  const float* Wv = (const float*)d_in[9];  const float* bv = (const float*)d_in[10];
  const float* Wo = (const float*)d_in[11]; const float* bo = (const float*)d_in[12];
  const float* gamma = (const float*)d_in[13];
  const float* beta  = (const float*)d_in[14];

  const int M = 512, MD = 1024;            // rows = B*L, model dim
  const long long SL = 524288;             // 512*1024 elements per buffer
  float* ws = (float*)d_ws;
  float* qp   = ws + 0 * SL;               // q projection  [512,1024] == [32,128,128]
  float* kp   = ws + 1 * SL;               // k projection
  float* vp   = ws + 2 * SL;               // v projection
  float* vn   = ws + 3 * SL;               // v + rel_v term
  float* ctx  = kp;                        // reuse: k dead after S GEMM
  float* opre = vp;                        // reuse: v dead after rel_v_update

  float* outp = (float*)d_out;             // LN output [512,1024]
  float* attn = (float*)d_out + SL;        // attn [32,128,128]

  dim3 blk(128);
  dim3 gProj(MD / 64, M / 64, 1);          // (16,8)
  dim3 gHead(2, 2, 32);                    // per-head 128x128x128
  dim3 gRel(128, 32);

  // 1-3: q/k/v projections (WMMA bf16)
  gemm_xwT_bf16<<<gProj, blk, 0, stream>>>(query, Wq, bq, nullptr, qp, M, MD, MD, 0, 0, 0);
  gemm_xwT_bf16<<<gProj, blk, 0, stream>>>(key,   Wk, bk, nullptr, kp, M, MD, MD, 0, 0, 0);
  gemm_xwT_bf16<<<gProj, blk, 0, stream>>>(value, Wv, bv, nullptr, vp, M, MD, MD, 0, 0, 0);

  // 4: S = q . k^T per head -> attn buffer
  gemm_xwT_bf16<<<gHead, blk, 0, stream>>>(qp, kp, nullptr, nullptr, attn,
                                           128, 128, 128, 16384, 16384, 16384);

  // 5: attn = softmax((S + q.relq)*0.25) in place (streams 268MB NT)
  rel_attn_softmax<<<gRel, blk, 0, stream>>>(qp, relq, attn);

  // 6: vn = v + v.relv (streams 268MB NT)
  rel_v_update<<<gRel, blk, 0, stream>>>(vp, relv, vn);

  // 7: context = attn . vn per head
  gemm_ab_bf16<<<gHead, blk, 0, stream>>>(attn, vn, ctx,
                                          128, 128, 128, 16384, 16384, 16384);

  // 8: out_pre = ctx . Wo^T + bo + residual(query)
  gemm_xwT_bf16<<<gProj, blk, 0, stream>>>(ctx, Wo, bo, query, opre, M, MD, MD, 0, 0, 0);

  // 9: LayerNorm -> d_out
  layernorm_1024<<<dim3(512), dim3(256), 0, stream>>>(opre, gamma, beta, outp);
}
